// CharLSTM_1297080123702
// MI455X (gfx1250) — compile-verified, matrix operand order checked by reading, not run on hardware
//
#include <hip/hip_runtime.h>
#include <stdint.h>

#define SEQ   8192
#define EMBD  1024
#define HID   2048
#define NCH   256
#define D4    (4*HID)      // 8192 gate rows
#define DIN   (EMBD+HID)   // 3072

typedef __attribute__((ext_vector_type(16))) __bf16 v16bf;
typedef __attribute__((ext_vector_type(8)))  __bf16 v8bf;
typedef __attribute__((ext_vector_type(8)))  float  v8f;

// ---------------- workspace layout (bytes) ----------------
#define ZX_OFF   0ull                                     // fp32 [SEQ, D4]      268 MB
#define HST_OFF  (ZX_OFF  + (size_t)SEQ*D4*4)             // bf16 [SEQ, HID]      33 MB
#define XBF_OFF  (HST_OFF + (size_t)SEQ*HID*2)            // bf16 [SEQ, EMBD]     16 MB
#define WXB_OFF  (XBF_OFF + (size_t)SEQ*EMBD*2)           // bf16 [D4, EMBD]      16 MB
#define WHB_OFF  (WXB_OFF + (size_t)D4*EMBD*2)            // bf16 [D4, HID]       33 MB
#define WOB_OFF  (WHB_OFF + (size_t)D4*HID*2)             // bf16 [NCH, HID]       1 MB
#define B4_OFF   (WOB_OFF + (size_t)NCH*HID*2)            // fp32 [D4]
#define HB0_OFF  (B4_OFF  + (size_t)D4*4)                 // fp32 [HID]
#define HB1_OFF  (HB0_OFF + (size_t)HID*4)                // fp32 [HID]
#define CB_OFF   (HB1_OFF + (size_t)HID*4)                // fp32 [HID]
#define CNT_OFF  (CB_OFF  + (size_t)HID*4)                // u32 barrier counter

__device__ __forceinline__ float bflo(unsigned u) {
  return __builtin_bit_cast(float, u << 16);
}
__device__ __forceinline__ float bfhi(unsigned u) {
  return __builtin_bit_cast(float, u & 0xffff0000u);
}
__device__ __forceinline__ float sigmoidf_(float x) {
  return 1.0f / (1.0f + __expf(-x));
}

// ---------------- prep kernels (fp32 -> bf16, gather, bias concat) ----------
__global__ __launch_bounds__(256) void k_prep_gates(
    const float* __restrict__ Wf, const float* __restrict__ Wi,
    const float* __restrict__ Wo, const float* __restrict__ Wc,
    const float* __restrict__ bf_, const float* __restrict__ bi_,
    const float* __restrict__ bo_, const float* __restrict__ bc_,
    __bf16* __restrict__ Wx, __bf16* __restrict__ Wh, float* __restrict__ b4) {
  const size_t total = (size_t)D4 * DIN;
  for (size_t idx = (size_t)blockIdx.x * blockDim.x + threadIdx.x; idx < total;
       idx += (size_t)gridDim.x * blockDim.x) {
    const int g = (int)(idx / DIN);       // gate-major row 0..8191
    const int c = (int)(idx % DIN);       // column 0..3071
    const int G = g >> 11;                // which gate (f,i,o,c)
    const int r = g & (HID - 1);          // row within gate
    const float* W = (G == 0) ? Wf : (G == 1) ? Wi : (G == 2) ? Wo : Wc;
    const float v = W[(size_t)r * DIN + c];
    if (c < EMBD) Wx[(size_t)g * EMBD + c] = (__bf16)v;
    else          Wh[(size_t)g * HID + (c - EMBD)] = (__bf16)v;
    if (idx < (size_t)D4) {
      const int gg = (int)idx, GG = gg >> 11, rr = gg & (HID - 1);
      const float* B = (GG == 0) ? bf_ : (GG == 1) ? bi_ : (GG == 2) ? bo_ : bc_;
      b4[gg] = B[rr];
    }
  }
}

__global__ __launch_bounds__(256) void k_prep_x(
    const int* __restrict__ seq, const float* __restrict__ emb,
    __bf16* __restrict__ X) {
  const size_t total = (size_t)SEQ * EMBD;
  for (size_t idx = (size_t)blockIdx.x * blockDim.x + threadIdx.x; idx < total;
       idx += (size_t)gridDim.x * blockDim.x) {
    const int t = (int)(idx >> 10);
    const int k = (int)(idx & (EMBD - 1));
    X[idx] = (__bf16)emb[(size_t)seq[t] * EMBD + k];
  }
}

__global__ __launch_bounds__(256) void k_prep_wout(
    const float* __restrict__ Wout, __bf16* __restrict__ Wob) {
  const size_t idx = (size_t)blockIdx.x * blockDim.x + threadIdx.x;
  if (idx < (size_t)NCH * HID) Wob[idx] = (__bf16)Wout[idx];
}

__global__ __launch_bounds__(256) void k_init(
    float* __restrict__ hb0, float* __restrict__ hb1, float* __restrict__ cb,
    unsigned* __restrict__ cnt) {
  const int idx = blockIdx.x * blockDim.x + threadIdx.x;
  if (idx < HID) { hb0[idx] = 0.f; hb1[idx] = 0.f; cb[idx] = 0.f; }
  if (idx < 64) cnt[idx] = 0u;
}

// ---------------- bf16 WMMA GEMM: C[M,N] = A[M,K] * B[N,K]^T + bias[N] ------
// Block tile 64(M) x 256(N), 8 waves arranged 2x4, wave tile 32x64.
__global__ __launch_bounds__(256) void k_gemm_bf16(
    float* __restrict__ C, const __bf16* __restrict__ A,
    const __bf16* __restrict__ B, const float* __restrict__ bias,
    int M, int N, int K) {
  const int tid  = threadIdx.x;
  const int lane = tid & 31;
  const int wave = tid >> 5;
  const int wm   = wave >> 2;               // 0..1
  const int wn   = wave & 3;                // 0..3
  const int m0   = blockIdx.x * 64  + wm * 32;
  const int n0   = blockIdx.y * 256 + wn * 64;
  const int lr    = lane & 15;              // row within 16-tile
  const int lhalf = lane >> 4;              // lanes 16..31 take +8 K-halves
  const int koff  = lhalf * 8;

  v8f acc[2][4];
#pragma unroll
  for (int i = 0; i < 2; ++i)
#pragma unroll
    for (int j = 0; j < 4; ++j) acc[i][j] = (v8f){0.f,0.f,0.f,0.f,0.f,0.f,0.f,0.f};

  for (int k0 = 0; k0 < K; k0 += 32) {
    v16bf a[2], b[4];
#pragma unroll
    for (int i = 0; i < 2; ++i) {
      const __bf16* p = A + (size_t)(m0 + i * 16 + lr) * K + k0 + koff;
      v8bf lo = *(const v8bf*)p;
      v8bf hi = *(const v8bf*)(p + 16);
#pragma unroll
      for (int q = 0; q < 8; ++q) { a[i][q] = lo[q]; a[i][q + 8] = hi[q]; }
    }
#pragma unroll
    for (int j = 0; j < 4; ++j) {
      const __bf16* p = B + (size_t)(n0 + j * 16 + lr) * K + k0 + koff;
      v8bf lo = *(const v8bf*)p;
      v8bf hi = *(const v8bf*)(p + 16);
#pragma unroll
      for (int q = 0; q < 8; ++q) { b[j][q] = lo[q]; b[j][q + 8] = hi[q]; }
    }
#pragma unroll
    for (int i = 0; i < 2; ++i)
#pragma unroll
      for (int j = 0; j < 4; ++j)
        acc[i][j] = __builtin_amdgcn_wmma_f32_16x16x32_bf16(
            false, a[i], false, b[j], (short)0, acc[i][j], false, false);
  }

  const int rowoff = lhalf * 8;             // D layout: lanes 16..31 hold rows 8..15
#pragma unroll
  for (int i = 0; i < 2; ++i)
#pragma unroll
    for (int j = 0; j < 4; ++j) {
      const int col = n0 + j * 16 + lr;
      const float bv = bias[col];           // bias always provided
      float* cp = C + (size_t)(m0 + i * 16 + rowoff) * N + col;
#pragma unroll
      for (int q = 0; q < 8; ++q) cp[(size_t)q * N] = acc[i][j][q] + bv;
    }
}

// ---------------- persistent recurrent kernel -------------------------------
// 128 WGs x 256 threads. WG `wg` owns hidden units [wg*16, wg*16+16), i.e. the
// 64 gate rows {G*2048 + wg*16 + jl}. h double-buffered; one grid barrier/step.
__global__ __launch_bounds__(256) void k_lstm(
    const float* __restrict__ Zx, const __bf16* __restrict__ Wh,
    float* __restrict__ hb0, float* __restrict__ hb1, float* __restrict__ cb,
    __bf16* __restrict__ Hst, unsigned* __restrict__ cnt) {
  __shared__ __align__(16) float h_lds[HID];
  __shared__ float zbuf[64];
  const int tid  = threadIdx.x;
  const int lane = tid & 31;
  const int wave = tid >> 5;        // 0..7, each wave does 8 gate rows
  const int wg   = blockIdx.x;
  const unsigned nwg = gridDim.x;
  const float4* h4 = (const float4*)h_lds;

  for (int t = 0; t < SEQ; ++t) {
    const float* hr = (t & 1) ? hb1 : hb0;
    float*       hw = (t & 1) ? hb0 : hb1;
    {
      const float4* hr4 = (const float4*)hr;
      float4* hl4 = (float4*)h_lds;
      for (int i = tid; i < HID / 4; i += 256) hl4[i] = hr4[i];
    }
    __syncthreads();

    for (int rl = wave * 8; rl < wave * 8 + 8; ++rl) {
      const int G  = rl >> 4;
      const int jl = rl & 15;
      const int g  = G * HID + wg * 16 + jl;
      // warm WGP cache for the next consecutive row (4 KB = 32 lanes x 128 B)
      __builtin_prefetch((const char*)(Wh + (size_t)(g + 1) * HID) + lane * 128,
                         0, 1);
      const uint4* wrow = (const uint4*)(Wh + (size_t)g * HID); // 256 uint4/row
      float acc = 0.f;
#pragma unroll
      for (int i = lane; i < HID / 8; i += 32) {   // 8 iters: b128 weight loads
        const uint4  w  = wrow[i];
        const float4 ha = h4[2 * i];
        const float4 hb = h4[2 * i + 1];
        acc += bflo(w.x) * ha.x + bfhi(w.x) * ha.y
             + bflo(w.y) * ha.z + bfhi(w.y) * ha.w
             + bflo(w.z) * hb.x + bfhi(w.z) * hb.y
             + bflo(w.w) * hb.z + bfhi(w.w) * hb.w;
      }
#pragma unroll
      for (int off = 16; off > 0; off >>= 1) acc += __shfl_xor(acc, off, 32);
      if (lane == 0) zbuf[rl] = acc;
    }
    __syncthreads();

    if (tid < 16) {
      const int j = wg * 16 + tid;
      const float* zx = Zx + (size_t)t * D4;          // b4 already fused here
      const float zf = zbuf[tid]      + zx[j];
      const float zi = zbuf[16 + tid] + zx[HID + j];
      const float zo = zbuf[32 + tid] + zx[2 * HID + j];
      const float zc = zbuf[48 + tid] + zx[3 * HID + j];
      const float f  = sigmoidf_(zf);
      const float ig = sigmoidf_(zi);
      const float o  = sigmoidf_(zo);
      const float ct = tanhf(zc);
      const float cn = f * cb[j] + ig * ct;
      const float hn = o * tanhf(cn);
      cb[j] = cn;
      hw[j] = hn;
      Hst[(size_t)t * HID + j] = (__bf16)hn;
    }

    // device-wide barrier: all 16*128 new h values visible before next step
    __syncthreads();
    if (tid == 0) {
      __threadfence();
      atomicAdd(cnt, 1u);
      const unsigned target = (unsigned)(t + 1) * nwg;
      while (atomicAdd(cnt, 0u) < target) __builtin_amdgcn_s_sleep(1);
      __threadfence();
    }
    __syncthreads();
  }
}

__global__ __launch_bounds__(256) void k_final(
    const float* __restrict__ hfin, const float* __restrict__ cfin,
    float* __restrict__ out) {
  const int idx = blockIdx.x * blockDim.x + threadIdx.x;
  if (idx < HID) {
    out[(size_t)SEQ * NCH + idx]       = hfin[idx];
    out[(size_t)SEQ * NCH + HID + idx] = cfin[idx];
  }
}

// ---------------- host launcher --------------------------------------------
extern "C" void kernel_launch(void* const* d_in, const int* in_sizes, int n_in,
                              void* d_out, int out_size, void* d_ws, size_t ws_size,
                              hipStream_t stream) {
  const int*   seq  = (const int*)d_in[0];
  const float* emb  = (const float*)d_in[1];
  const float* Wf   = (const float*)d_in[2];
  const float* bf_  = (const float*)d_in[3];
  const float* Wi   = (const float*)d_in[4];
  const float* bi_  = (const float*)d_in[5];
  const float* Wo   = (const float*)d_in[6];
  const float* bo_  = (const float*)d_in[7];
  const float* Wc   = (const float*)d_in[8];
  const float* bc_  = (const float*)d_in[9];
  const float* Wout = (const float*)d_in[10];
  const float* bout = (const float*)d_in[11];

  char* ws = (char*)d_ws;
  float*    Zx  = (float*)(ws + ZX_OFF);
  __bf16*   Hst = (__bf16*)(ws + HST_OFF);
  __bf16*   Xbf = (__bf16*)(ws + XBF_OFF);
  __bf16*   Wxb = (__bf16*)(ws + WXB_OFF);
  __bf16*   Whb = (__bf16*)(ws + WHB_OFF);
  __bf16*   Wob = (__bf16*)(ws + WOB_OFF);
  float*    b4  = (float*)(ws + B4_OFF);
  float*    hb0 = (float*)(ws + HB0_OFF);
  float*    hb1 = (float*)(ws + HB1_OFF);
  float*    cb  = (float*)(ws + CB_OFF);
  unsigned* cnt = (unsigned*)(ws + CNT_OFF);
  float*    out = (float*)d_out;

  // phase 0: precision conversion / gather / init
  k_prep_gates<<<8192, 256, 0, stream>>>(Wf, Wi, Wo, Wc, bf_, bi_, bo_, bc_,
                                         Wxb, Whb, b4);
  k_prep_x<<<4096, 256, 0, stream>>>(seq, emb, Xbf);
  k_prep_wout<<<(NCH * HID) / 256, 256, 0, stream>>>(Wout, Wob);
  k_init<<<8, 256, 0, stream>>>(hb0, hb1, cb, cnt);

  // phase 1: Zx[t,g] = sum_k X[t,k]*Wx[g,k] + b4[g]   (WMMA bf16 GEMM)
  {
    dim3 grid(SEQ / 64, D4 / 256);
    k_gemm_bf16<<<grid, 256, 0, stream>>>(Zx, Xbf, Wxb, b4, SEQ, D4, EMBD);
  }

  // phase 2: sequential recurrence (persistent kernel, 1 grid barrier/step)
  k_lstm<<<128, 256, 0, stream>>>(Zx, Whb, hb0, hb1, cb, Hst, cnt);

  // phase 3: Out[t,n] = sum_k H[t,k]*Wout[n,k] + bout[n]  (WMMA bf16 GEMM)
  {
    dim3 grid(SEQ / 64, NCH / 256);
    k_gemm_bf16<<<grid, 256, 0, stream>>>(out, Hst, Wob, bout, SEQ, NCH, HID);
  }

  // phase 4: final h (ends in hb0 after 8192 steps) and c to output tail
  k_final<<<8, 256, 0, stream>>>(hb0, cb, out);
}
